// _DenseGATLayer_78589311582484
// MI455X (gfx1250) — matrix operand order repaired and checked by reading, not compiled, not run on hardware
//
#include <hip/hip_runtime.h>

// ---- problem sizes (DenseGAT) ----
#define NN   8192      // nodes
#define KC   256       // IN_C
#define OC   256       // OUT_C
#define NH   4         // heads
#define NEG  0.2f      // leaky slope

typedef __bf16 bf16;
typedef __bf16 v16bf __attribute__((ext_vector_type(16)));
typedef __bf16 v8bf  __attribute__((ext_vector_type(8)));
typedef __bf16 v4bf  __attribute__((ext_vector_type(4)));
typedef float  v8f   __attribute__((ext_vector_type(8)));
typedef float  v4f   __attribute__((ext_vector_type(4)));
typedef unsigned u32x4 __attribute__((ext_vector_type(4)));
typedef unsigned u32x8 __attribute__((ext_vector_type(8)));

static __device__ __forceinline__ v16bf cat16(v8bf lo, v8bf hi) {
  v16bf r;
#pragma unroll
  for (int i = 0; i < 8; ++i) { r[i] = lo[i]; r[i + 8] = hi[i]; }
  return r;
}

static __device__ __forceinline__ void ld8(float* d, const float* s) {
  v4f a = *(const v4f*)s;
  v4f b = *(const v4f*)(s + 4);
#pragma unroll
  for (int i = 0; i < 4; ++i) { d[i] = a[i]; d[i + 4] = b[i]; }
}

// monotone f32 <-> int encoding so atomicMax(int) == float max
static __device__ __forceinline__ int fenc(float x) {
  int i = __float_as_int(x);
  return (i >= 0) ? i : (i ^ 0x7fffffff);
}
static __device__ __forceinline__ float fdec(int i) {
  return __int_as_float((i >= 0) ? i : (i ^ 0x7fffffff));
}

// ---- TDM: tensor_load_to_lds with a 2-group D# (<=2D tile, optional LDS pad) ----
// Descriptor packing per CDNA5 ISA ch.8 (group0 128b, group1 256b).
static __device__ __forceinline__ void tdm_load(const void* lds_ptr, const void* gptr,
                                                unsigned dsz_log2,   // 0=1B,1=2B,2=4B,3=8B
                                                unsigned tile0, unsigned tile1,
                                                unsigned long long stride0_elems,
                                                unsigned tdim0, unsigned tdim1,
                                                unsigned pad_interval, unsigned pad_amount,
                                                unsigned pad_en) {
  unsigned long long ga = (unsigned long long)gptr;
  u32x4 g0;
  g0[0] = 1u;                                               // count=1, user D#
  g0[1] = (unsigned)(unsigned long long)lds_ptr;            // LDS byte offset (addr[31:0])
  g0[2] = (unsigned)ga;                                     // global_addr[31:0]
  g0[3] = (unsigned)((ga >> 32) & 0x01ffffffu) | (2u << 30);// global_addr[56:32], type=2
  u32x8 g1;
  g1[0] = (dsz_log2 << 16) | (pad_en << 20) | (pad_interval << 22) | (pad_amount << 25);
  g1[1] = (tdim0 & 0xffffu) << 16;                          // tensor_dim0[15:0]  @ bits 63:48
  g1[2] = ((tdim0 >> 16) & 0xffffu) | ((tdim1 & 0xffffu) << 16);
  g1[3] = ((tdim1 >> 16) & 0xffffu) | ((tile0 & 0xffffu) << 16); // tile_dim0 @ 127:112
  g1[4] = tile1 & 0xffffu;                                  // tile_dim1 (tile_dim2=0)
  g1[5] = (unsigned)stride0_elems;                          // tensor_dim0_stride[31:0]
  g1[6] = (unsigned)((stride0_elems >> 32) & 0xffffu);      // stride[47:32]
  g1[7] = 0u;
  asm volatile("tensor_load_to_lds %0, %1" :: "s"(g0), "s"(g1) : "memory");
}

static __device__ __forceinline__ void async_b32(const void* lds_ptr, const void* gptr) {
  unsigned l = (unsigned)(unsigned long long)lds_ptr;       // LDS byte offset
  asm volatile("global_load_async_to_lds_b32 %0, %1, off" :: "v"(l), "v"(gptr) : "memory");
}

// ---------------- kernel 1: x (f32) -> bf16 ----------------
__global__ void __launch_bounds__(256) k_cvt_x(const float* __restrict__ x,
                                               bf16* __restrict__ xb) {
  int i = (blockIdx.x * 256 + threadIdx.x) * 4;   // 2048 blocks cover 2M elems
  v4f v = *(const v4f*)(x + i);
  v4bf o;
#pragma unroll
  for (int r = 0; r < 4; ++r) o[r] = (bf16)v[r];
  *(v4bf*)(xb + i) = o;
}

// ---------------- kernel 2: W -> W^T bf16, init maxdst ----------------
__global__ void __launch_bounds__(256) k_cvt_w(const float* __restrict__ W,
                                               bf16* __restrict__ wtb,
                                               int* __restrict__ mx) {
  int idx = blockIdx.x * 256 + threadIdx.x;       // 65536
  int f = idx >> 8, k = idx & 255;
  wtb[idx] = (bf16)W[(size_t)k * OC + f];         // wtb[f][k] = W[k][f]
  if (idx < NH) mx[idx] = (int)0x80000000;
}

// ---------------- kernel 3: h = x @ W  (bf16 WMMA) ----------------
// Output stored transposed+tiled: hT[kb][f][j&31], kb = j>>5  (bf16)
__global__ void __launch_bounds__(128) k_gemm_h(const bf16* __restrict__ xb,
                                                const bf16* __restrict__ wtb,
                                                bf16* __restrict__ hT) {
  int bid  = blockIdx.x;                 // 2048 = 512 j-tiles * 4 f-groups
  int jt   = bid >> 2, fg = bid & 3;
  int wave = threadIdx.x >> 5, lane = threadIdx.x & 31;
  int ln = lane & 15, hf = lane >> 4;
  int j0 = jt * 16, f0 = (fg * 4 + wave) * 16;

  const bf16* arow = xb  + (size_t)(j0 + ln) * KC;   // A row m = ln
  const bf16* brow = wtb + (size_t)(f0 + ln) * KC;   // B col n = ln (W^T row)
  v8f acc = {};
#pragma unroll
  for (int k0 = 0; k0 < KC; k0 += 32) {
    // A layout: lanes<16: K {0-7,16-23}; lanes>=16: K {8-15,24-31}
    v16bf A = cat16(*(const v8bf*)(arow + k0 + hf * 8),
                    *(const v8bf*)(arow + k0 + 16 + hf * 8));
    // B layout: lanes<16: K {0-7,16-23}; lanes>=16: K {8-15,24-31}
    v16bf B = cat16(*(const v8bf*)(brow + k0 + hf * 8),
                    *(const v8bf*)(brow + k0 + 16 + hf * 8));
    acc = __builtin_amdgcn_wmma_f32_16x16x32_bf16(false, A, false, B,
                                                  (short)0, acc, false, false);
  }
  // C layout: lane holds column n=ln, rows 8*hf + r -> 8 consecutive j
  int j  = j0 + 8 * hf;
  int kb = j >> 5, jj = j & 31;
  v8bf ov;
#pragma unroll
  for (int r = 0; r < 8; ++r) ov[r] = (bf16)acc[r];
  *(v8bf*)(hT + (size_t)kb * (OC * 32) + (size_t)(f0 + ln) * 32 + jj) = ov;
}

// ---------------- kernel 4: e_src/e_dst dots + per-head max(e_dst) ----------------
__global__ void __launch_bounds__(256) k_edge(const bf16* __restrict__ hT,
                                              const float* __restrict__ asrc,
                                              const float* __restrict__ adst,
                                              float* __restrict__ es,
                                              float* __restrict__ ed,
                                              int* __restrict__ mx) {
  int n = blockIdx.x * 256 + threadIdx.x;   // 32 blocks -> 8192 nodes
  const bf16* hp = hT + (size_t)(n >> 5) * (OC * 32) + (n & 31);
#pragma unroll
  for (int h = 0; h < NH; ++h) {
    float as = 0.f, ad = 0.f;
    for (int f2 = 0; f2 < 64; ++f2) {
      int f = h * 64 + f2;
      float v = (float)hp[(size_t)f * 32];   // lanes: consecutive n -> coalesced
      as += v * asrc[f];
      ad += v * adst[f];
    }
    es[n * NH + h] = as;
    ed[n * NH + h] = ad;
    atomicMax(&mx[h], fenc(ad));
  }
}

// ---------------- kernel 5: softmax factor tables ----------------
__global__ void __launch_bounds__(256) k_coeff(const float* __restrict__ es,
                                               const float* __restrict__ ed,
                                               const int* __restrict__ mx,
                                               float* __restrict__ c1, float* __restrict__ c2,
                                               float* __restrict__ d1, float* __restrict__ d2) {
  int idx = blockIdx.x * 256 + threadIdx.x;    // 128 blocks -> 32768
  int h = idx & (NH - 1);
  float md   = fdec(mx[h]);
  float e_s  = es[idx], e_d = ed[idx];
  float smax = e_s + md;
  float M    = (smax > 0.f) ? smax : NEG * smax;   // stabilizer >= all leaky logits
  c1[idx] = expf(e_s + md - M);
  c2[idx] = expf(NEG * (e_s + md) - M);
  d1[idx] = expf(e_d - md);
  d2[idx] = expf(NEG * (e_d - md));
}

// ---------------- kernel 6: fused masked-softmax attention + WMMA aggregate ----------------
// 512 blocks (16 output rows each) x 128 threads (wave = head).
// 3-stage TDM/async pipeline, one barrier per j-block.
__global__ void __launch_bounds__(128) k_attn(const float* __restrict__ P,
                                              const bf16* __restrict__ hT,
                                              const float* __restrict__ es,
                                              const float* __restrict__ edg,
                                              const float* __restrict__ c1g,
                                              const float* __restrict__ c2g,
                                              const float* __restrict__ d1g,
                                              const float* __restrict__ d2g,
                                              float* __restrict__ out) {
  __shared__ __align__(16) bf16  ldsH[3][OC * 40];   // h tile [256 f][32 j], TDM-padded rows (40)
  __shared__ __align__(16) float ldsP[3][16 * 36];   // P tile [16 i][32 j], TDM-padded rows (36)
  __shared__ __align__(16) float ldsED[3][3 * 128];  // {e_dst,d1,d2}[head][32 j]
  __shared__ float ldsDen[64];                       // denom[head][16 rows]

  const int tid  = threadIdx.x;
  const int hd   = tid >> 5;          // head (wave id)
  const int lane = tid & 31;
  const int ln   = lane & 15;
  const int hf   = lane >> 4;
  const int c0   = hf * 8;            // A-matrix k-chunk base per half-wave
  const int i0   = blockIdx.x * 16;
  const int ek   = lane;              // ED staging: j within block

  const float esrc = es [(i0 + ln) * NH + hd];
  const float c1v  = c1g[(i0 + ln) * NH + hd];
  const float c2v  = c2g[(i0 + ln) * NH + hd];

  v8f acc[4] = {{}, {}, {}, {}};      // 16 rows x 64 f (4 N-tiles)
  float wsum = 0.f;

  auto stage = [&](int b, int jb) {
    if (hd == 0) {
      // 16KB contiguous h tile; pad 16B after every 64B -> LDS rows of 40 bf16
      tdm_load(&ldsH[b][0], hT + (size_t)(jb >> 5) * (OC * 32),
               /*8B*/3, /*tile*/2048, 1, /*stride*/2048, /*tensor*/2048, 1,
               /*padi 64B*/3, /*pada 16B*/3, 1);
      // P tile 16x32 f32, row stride NN; pad 16B after every 128B -> LDS rows of 36 f32
      tdm_load(&ldsP[b][0], P + (size_t)i0 * NN + jb,
               /*4B*/2, /*tile*/32, 16, /*stride*/NN, /*tensor*/NN, NN,
               /*padi 128B*/4, /*pada 16B*/3, 1);
    }
    // per-head transposed e_dst/d1/d2 rows: staged and read by the same wave
    int gi = (jb + ek) * NH + hd;
    async_b32(&ldsED[b][      hd * 32 + ek], edg + gi);
    async_b32(&ldsED[b][128 + hd * 32 + ek], d1g + gi);
    async_b32(&ldsED[b][256 + hd * 32 + ek], d2g + gi);
  };

  stage(0, 0);        // prologue: two stages in flight
  stage(1, 32);

  for (int it = 0; it < NN / 32; ++it) {
    const int jb = it * 32;
    const int b  = it % 3;
    // loads complete in order: waiting "<= next stage's count" => current tile resident
    if (jb + 32 < NN) {
      if (hd == 0) __builtin_amdgcn_s_wait_tensorcnt(2);
      asm volatile("s_wait_asynccnt 0x3" ::: "memory");
    } else {
      if (hd == 0) __builtin_amdgcn_s_wait_tensorcnt(0);
      asm volatile("s_wait_asynccnt 0x0" ::: "memory");
    }
    __syncthreads();  // publish wave-0 TDM tiles; also WAR fence for re-staged buffer

    // ---- build A (attention weights, bf16) directly in WMMA A layout ----
    // lane: row m = ln, k-chunks {c0..c0+7} and {c0+16..c0+23}
    float pv[16], ev[16], dav[16], dbv[16];
    const float* prow = &ldsP[b][ln * 36];
    ld8(pv,      prow + c0);                   ld8(pv + 8,  prow + c0 + 16);
    ld8(ev,      &ldsED[b][      hd*32 + c0]); ld8(ev + 8,  &ldsED[b][      hd*32 + c0 + 16]);
    ld8(dav,     &ldsED[b][128 + hd*32 + c0]); ld8(dav + 8, &ldsED[b][128 + hd*32 + c0 + 16]);
    ld8(dbv,     &ldsED[b][256 + hd*32 + c0]); ld8(dbv + 8, &ldsED[b][256 + hd*32 + c0 + 16]);

    v16bf A;
#pragma unroll
    for (int u = 0; u < 16; ++u) {
      float p = pv[u];
      float s = esrc + ev[u];                          // pre-leaky logit
      float w = (s > 0.f) ? (c1v * dav[u]) : (c2v * dbv[u]);  // exp(leaky(s)-M)
      w = (p > 0.f) ? w : 0.f;                         // adjacency mask
      wsum += w;                                       // softmax denominator
      A[u] = (bf16)(w * p);                            // attn numerator weight * P
    }
    // ---- 4 WMMAs: out[16 x 64] += A[16 x 32j] * h[32j x 16f] ----
#pragma unroll
    for (int nt = 0; nt < 4; ++nt) {
      const bf16* bp = &ldsH[b][(hd * 64 + nt * 16 + ln) * 40 + hf * 8];
      v16bf B = cat16(*(const v8bf*)bp, *(const v8bf*)(bp + 16));
      acc[nt] = __builtin_amdgcn_wmma_f32_16x16x32_bf16(false, A, false, B,
                                                        (short)0, acc[nt], false, false);
    }

    // stage tile it+2 into buffer last read at it-1 (safe: we passed barrier_it,
    // which every wave reached only after finishing compute it-1)
    if (jb + 64 < NN) stage((it + 2) % 3, jb + 64);
  }

  // reduce denominator: lanes l and l+16 cover complementary k-chunks of row ln
  wsum += __shfl_xor(wsum, 16, 32);
  __syncthreads();
  if (hf == 0) ldsDen[hd * 16 + ln] = wsum;
  __syncthreads();

  const int rbase = hf * 8;           // C layout: lane holds rows rbase..rbase+7
#pragma unroll
  for (int r = 0; r < 8; ++r) {
    float inv = __builtin_amdgcn_rcpf(ldsDen[hd * 16 + rbase + r]);
    int row = i0 + rbase + r;
#pragma unroll
    for (int nt = 0; nt < 4; ++nt)
      out[(size_t)row * OC + hd * 64 + nt * 16 + ln] = acc[nt][r] * inv;
  }
}

// ---------------- host side ----------------
extern "C" void kernel_launch(void* const* d_in, const int* in_sizes, int n_in,
                              void* d_out, int out_size, void* d_ws, size_t ws_size,
                              hipStream_t stream) {
  (void)in_sizes; (void)n_in; (void)out_size; (void)ws_size;
  const float* x    = (const float*)d_in[0];
  const float* P    = (const float*)d_in[1];
  const float* W    = (const float*)d_in[2];
  const float* asrc = (const float*)d_in[3];   // [4][64] flat == index f
  const float* adst = (const float*)d_in[4];
  float* out = (float*)d_out;

  char* ws = (char*)d_ws;                      // ~9.3 MB total
  bf16*  xb  = (bf16*)(ws);                    // 4,194,304
  bf16*  wtb = (bf16*)(ws + 4194304);          //   131,072
  bf16*  hT  = (bf16*)(ws + 4325376);          // 4,194,304
  float* es  = (float*)(ws + 8519680);         //   131,072
  float* ed  = (float*)(ws + 8650752);
  float* c1  = (float*)(ws + 8781824);
  float* c2  = (float*)(ws + 8912896);
  float* d1  = (float*)(ws + 9043968);
  float* d2  = (float*)(ws + 9175040);
  int*   mx  = (int*)  (ws + 9306112);

  k_cvt_x <<<2048, 256, 0, stream>>>(x, xb);
  k_cvt_w <<< 256, 256, 0, stream>>>(W, wtb, mx);
  k_gemm_h<<<2048, 128, 0, stream>>>(xb, wtb, hT);
  k_edge  <<<  32, 256, 0, stream>>>(hT, asrc, adst, es, ed, mx);
  k_coeff <<< 128, 256, 0, stream>>>(es, ed, mx, c1, c2, d1, d2);
  k_attn  <<< 512, 128, 0, stream>>>(P, hT, es, ed, c1, c2, d1, d2, out);
}
